// EncoderTransformer_28905129902637
// MI455X (gfx1250) — compile-verified, matrix operand order checked by reading, not run on hardware
//
#include <hip/hip_runtime.h>
#include <hip/hip_bf16.h>

#define D_  768
#define H_  12
#define HD_ 64
#define T_  2048
#define L_  4
#define M_  64
#define S_  (M_ + T_)     // 2112
#define FF_ (4 * D_)      // 3072

typedef __attribute__((ext_vector_type(16))) __bf16 v16bf;
typedef __attribute__((ext_vector_type(8)))  __bf16 v8bf;
typedef __attribute__((ext_vector_type(8)))  float  v8f;

static __device__ __forceinline__ __bf16 f2bf(float f) {
  unsigned u = __float_as_uint(f);
  u += 0x7fffu + ((u >> 16) & 1u);           // round-to-nearest-even
  unsigned short s = (unsigned short)(u >> 16);
  return *reinterpret_cast<__bf16*>(&s);
}

// ---------------- elementwise / setup kernels ----------------

__global__ void cast_f32_bf16(const float* __restrict__ src, __bf16* __restrict__ dst, int n) {
  int i = blockIdx.x * blockDim.x + threadIdx.x;
  if (i < n) dst[i] = f2bf(src[i]);
}

__global__ void embed_kernel(const int* __restrict__ tokens, const float* __restrict__ emb,
                             const float* __restrict__ pos, const float* __restrict__ mem,
                             float* __restrict__ h) {
  int i = blockIdx.x * blockDim.x + threadIdx.x;
  if (i >= S_ * D_) return;
  int s = i / D_, d = i - s * D_;
  float v;
  if (s < M_) {
    v = mem[s * D_ + d];
  } else {
    int t = s - M_;
    v = emb[(size_t)tokens[t] * D_ + d] + pos[(size_t)t * D_ + d];
  }
  h[i] = v;
}

// ---------------- layernorm kernels ----------------

__global__ void ln_bf16_kernel(const float* __restrict__ x, __bf16* __restrict__ out,
                               const float* __restrict__ sc, const float* __restrict__ bi) {
  __shared__ float s1[256], s2[256];
  int row = blockIdx.x, tid = threadIdx.x;
  const float* xr = x + (size_t)row * D_;
  float a = 0.f, b = 0.f;
  for (int d = tid; d < D_; d += 256) { float v = xr[d]; a += v; b += v * v; }
  s1[tid] = a; s2[tid] = b; __syncthreads();
  for (int off = 128; off > 0; off >>= 1) {
    if (tid < off) { s1[tid] += s1[tid + off]; s2[tid] += s2[tid + off]; }
    __syncthreads();
  }
  float mu  = s1[0] / (float)D_;
  float var = s2[0] / (float)D_ - mu * mu;
  float inv = rsqrtf(var + 1e-5f);
  for (int d = tid; d < D_; d += 256)
    out[(size_t)row * D_ + d] = f2bf((xr[d] - mu) * inv * sc[d] + bi[d]);
}

__global__ void ln_f32out_kernel(const float* __restrict__ x, float* __restrict__ out,
                                 const float* __restrict__ sc, const float* __restrict__ bi) {
  __shared__ float s1[256], s2[256];
  int row = blockIdx.x, tid = threadIdx.x;
  const float* xr = x + (size_t)row * D_;
  float a = 0.f, b = 0.f;
  for (int d = tid; d < D_; d += 256) { float v = xr[d]; a += v; b += v * v; }
  s1[tid] = a; s2[tid] = b; __syncthreads();
  for (int off = 128; off > 0; off >>= 1) {
    if (tid < off) { s1[tid] += s1[tid + off]; s2[tid] += s2[tid + off]; }
    __syncthreads();
  }
  float mu  = s1[0] / (float)D_;
  float var = s2[0] / (float)D_ - mu * mu;
  float inv = rsqrtf(var + 1e-5f);
  for (int d = tid; d < D_; d += 256)
    out[(size_t)row * D_ + d] = (xr[d] - mu) * inv * sc[d] + bi[d];
}

// ---------------- causal softmax (one block per query row) ----------------

__global__ void softmax_causal_kernel(const float* __restrict__ sc, __bf16* __restrict__ P) {
  __shared__ float red[256];
  int q = blockIdx.x, tid = threadIdx.x;
  const float scale = 0.125f;                 // 1/sqrt(64)
  const float* row = sc + (size_t)q * S_;
  float mx = -3.4e38f;
  for (int k = tid; k <= q; k += 256) mx = fmaxf(mx, row[k] * scale);
  red[tid] = mx; __syncthreads();
  for (int off = 128; off > 0; off >>= 1) {
    if (tid < off) red[tid] = fmaxf(red[tid], red[tid + off]);
    __syncthreads();
  }
  mx = red[0]; __syncthreads();
  float sum = 0.f;
  for (int k = tid; k <= q; k += 256) sum += __expf(row[k] * scale - mx);
  red[tid] = sum; __syncthreads();
  for (int off = 128; off > 0; off >>= 1) {
    if (tid < off) red[tid] += red[tid + off];
    __syncthreads();
  }
  float inv = 1.f / red[0];
  __bf16* pr = P + (size_t)q * S_;
  for (int k = tid; k < S_; k += 256)
    pr[k] = (k <= q) ? f2bf(__expf(row[k] * scale - mx) * inv) : f2bf(0.f);
}

// ---------------- generic WMMA GEMM: C[M,N] = A[M,K] * B[N,K]^T ----------------
// Optionally batched over blockIdx.z with element strides on A, B, and output.

#define EPI_F32        0
#define EPI_BF16       1
#define EPI_BF16_GELU  2
#define EPI_F32_RES    3
#define EPI_BF16_TRANS 4

// One wave computes a 32(M) x 64(N) tile; grid = (N/64, M/32, Z).
// Per K-step: 4 A b128-loads + 8 B b128-loads feed 8 WMMAs (each B fragment
// reused by two WMMAs from registers).
__launch_bounds__(32)
__global__ void gemm_wmma_kernel(const __bf16* __restrict__ A, int lda,
                                 const __bf16* __restrict__ B, int ldb,
                                 float* __restrict__ outF, __bf16* __restrict__ outB, int ldc,
                                 const float* __restrict__ bias,
                                 const float* __restrict__ res,
                                 int K, int mode, int causal,
                                 size_t zStrideA, size_t zStrideB, size_t zStrideO) {
  int nb = blockIdx.x * 64;
  int mb = blockIdx.y * 32;
  if (causal && nb > mb + 31) return;         // fully masked score block
  size_t z = blockIdx.z;
  A += z * zStrideA;
  B += z * zStrideB;
  if (outF) outF += z * zStrideO;
  if (outB) outB += z * zStrideO;

  int lane = threadIdx.x;
  int lr = lane & 15;
  int hi = lane >> 4;                          // 0: K {0..7,16..23}; 1: K {8..15,24..31}
  const __bf16* arow0 = A + (size_t)(mb + lr) * lda;
  const __bf16* arow1 = A + (size_t)(mb + 16 + lr) * lda;
  v8f c00 = {}, c01 = {}, c02 = {}, c03 = {};
  v8f c10 = {}, c11 = {}, c12 = {}, c13 = {};
  for (int k0 = 0; k0 < K; k0 += 32) {
    int kb = k0 + hi * 8;
    union { v16bf v; v8bf h[2]; } a0, a1, b;
    a0.h[0] = *(const v8bf*)(arow0 + kb);
    a0.h[1] = *(const v8bf*)(arow0 + kb + 16);
    a1.h[0] = *(const v8bf*)(arow1 + kb);
    a1.h[1] = *(const v8bf*)(arow1 + kb + 16);

    const __bf16* br = B + (size_t)(nb + lr) * ldb + kb;
    b.h[0] = *(const v8bf*)(br);
    b.h[1] = *(const v8bf*)(br + 16);
    c00 = __builtin_amdgcn_wmma_f32_16x16x32_bf16(false, a0.v, false, b.v, (short)0, c00, false, false);
    c10 = __builtin_amdgcn_wmma_f32_16x16x32_bf16(false, a1.v, false, b.v, (short)0, c10, false, false);

    br = B + (size_t)(nb + 16 + lr) * ldb + kb;
    b.h[0] = *(const v8bf*)(br);
    b.h[1] = *(const v8bf*)(br + 16);
    c01 = __builtin_amdgcn_wmma_f32_16x16x32_bf16(false, a0.v, false, b.v, (short)0, c01, false, false);
    c11 = __builtin_amdgcn_wmma_f32_16x16x32_bf16(false, a1.v, false, b.v, (short)0, c11, false, false);

    br = B + (size_t)(nb + 32 + lr) * ldb + kb;
    b.h[0] = *(const v8bf*)(br);
    b.h[1] = *(const v8bf*)(br + 16);
    c02 = __builtin_amdgcn_wmma_f32_16x16x32_bf16(false, a0.v, false, b.v, (short)0, c02, false, false);
    c12 = __builtin_amdgcn_wmma_f32_16x16x32_bf16(false, a1.v, false, b.v, (short)0, c12, false, false);

    br = B + (size_t)(nb + 48 + lr) * ldb + kb;
    b.h[0] = *(const v8bf*)(br);
    b.h[1] = *(const v8bf*)(br + 16);
    c03 = __builtin_amdgcn_wmma_f32_16x16x32_bf16(false, a0.v, false, b.v, (short)0, c03, false, false);
    c13 = __builtin_amdgcn_wmma_f32_16x16x32_bf16(false, a1.v, false, b.v, (short)0, c13, false, false);
  }
  v8f cc[2][4] = {{c00, c01, c02, c03}, {c10, c11, c12, c13}};
#pragma unroll
  for (int mi = 0; mi < 2; ++mi) {
#pragma unroll
    for (int nf = 0; nf < 4; ++nf) {
#pragma unroll
      for (int r = 0; r < 8; ++r) {
        int m = mb + mi * 16 + r + hi * 8;     // C layout: lanes 16-31 hold rows +8..+15
        int col = nb + nf * 16 + lr;
        float acc = cc[mi][nf][r];
        if (mode == EPI_F32) {
          outF[(size_t)m * ldc + col] = acc;
        } else if (mode == EPI_BF16) {
          outB[(size_t)m * ldc + col] = f2bf(acc);
        } else if (mode == EPI_BF16_GELU) {
          float v = acc + (bias ? bias[col] : 0.f);
          float g = 0.5f * v * (1.f + tanhf(0.7978845608028654f * (v + 0.044715f * v * v * v)));
          outB[(size_t)m * ldc + col] = f2bf(g);
        } else if (mode == EPI_F32_RES) {
          float v = acc + (bias ? bias[col] : 0.f) + res[(size_t)m * ldc + col];
          outF[(size_t)m * ldc + col] = v;
        } else {                                // EPI_BF16_TRANS: ldc = transposed leading dim
          outB[(size_t)col * ldc + m] = f2bf(acc);
        }
      }
    }
  }
}

// ---------------- host orchestration ----------------

extern "C" void kernel_launch(void* const* d_in, const int* in_sizes, int n_in,
                              void* d_out, int out_size, void* d_ws, size_t ws_size,
                              hipStream_t stream) {
  (void)in_sizes; (void)n_in; (void)out_size; (void)ws_size;
  const int*   tokens = (const int*)  d_in[0];
  const float* emb    = (const float*)d_in[1];
  const float* pos    = (const float*)d_in[2];
  const float* mem    = (const float*)d_in[3];
  const float* ln1_s  = (const float*)d_in[4];
  const float* ln1_b  = (const float*)d_in[5];
  const float* wq     = (const float*)d_in[6];
  const float* wk     = (const float*)d_in[7];
  const float* wv     = (const float*)d_in[8];
  const float* wo     = (const float*)d_in[9];
  const float* ln2_s  = (const float*)d_in[10];
  const float* ln2_b  = (const float*)d_in[11];
  const float* w1     = (const float*)d_in[12];
  const float* b1     = (const float*)d_in[13];
  const float* w2     = (const float*)d_in[14];
  const float* b2     = (const float*)d_in[15];
  const float* lnm_s  = (const float*)d_in[16];
  const float* lnm_b  = (const float*)d_in[17];
  float* out = (float*)d_out;

  char* ws = (char*)d_ws;
  size_t off = 0;
  auto alloc = [&](size_t bytes) -> void* {
    void* p = ws + off;
    off = (off + bytes + 255) & ~(size_t)255;
    return p;
  };

  const size_t nDD  = (size_t)L_ * D_ * D_;
  const size_t nFD  = (size_t)L_ * FF_ * D_;
  __bf16* wq_bf  = (__bf16*)alloc(nDD * 2);
  __bf16* wk_bf  = (__bf16*)alloc(nDD * 2);
  __bf16* wv_bf  = (__bf16*)alloc(nDD * 2);
  __bf16* wo_bf  = (__bf16*)alloc(nDD * 2);
  __bf16* w1_bf  = (__bf16*)alloc(nFD * 2);
  __bf16* w2_bf  = (__bf16*)alloc(nFD * 2);
  float*  h      = (float*) alloc((size_t)S_ * D_ * 4);
  __bf16* nrm    = (__bf16*)alloc((size_t)S_ * D_ * 2);
  __bf16* qb     = (__bf16*)alloc((size_t)S_ * D_ * 2);
  __bf16* kb     = (__bf16*)alloc((size_t)S_ * D_ * 2);
  __bf16* vT     = (__bf16*)alloc((size_t)D_ * S_ * 2);        // [D, S]
  __bf16* ob     = (__bf16*)alloc((size_t)S_ * D_ * 2);
  float*  scores = (float*) alloc((size_t)S_ * S_ * 4);        // one head at a time
  __bf16* P      = (__bf16*)alloc((size_t)H_ * S_ * S_ * 2);   // per-head probs
  __bf16* act1   = (__bf16*)alloc((size_t)S_ * FF_ * 2);

  auto gemm = [&](const __bf16* A, int lda, const __bf16* B, int ldb,
                  float* oF, __bf16* oB, int ldc, const float* bias, const float* res,
                  int Mm, int Nn, int Kk, int mode, int causal,
                  int Zn, size_t sA, size_t sB, size_t sO) {
    dim3 g(Nn / 64, Mm / 32, Zn);
    gemm_wmma_kernel<<<g, dim3(32), 0, stream>>>(A, lda, B, ldb, oF, oB, ldc,
                                                 bias, res, Kk, mode, causal,
                                                 sA, sB, sO);
  };

  // Weight conversion f32 -> bf16 (amortized; weights are the HBM-heavy part).
  {
    int n = (int)nDD;
    int blocks = (n + 255) / 256;
    cast_f32_bf16<<<blocks, 256, 0, stream>>>(wq, wq_bf, n);
    cast_f32_bf16<<<blocks, 256, 0, stream>>>(wk, wk_bf, n);
    cast_f32_bf16<<<blocks, 256, 0, stream>>>(wv, wv_bf, n);
    cast_f32_bf16<<<blocks, 256, 0, stream>>>(wo, wo_bf, n);
    int nf = (int)nFD;
    int blocksF = (nf + 255) / 256;
    cast_f32_bf16<<<blocksF, 256, 0, stream>>>(w1, w1_bf, nf);
    cast_f32_bf16<<<blocksF, 256, 0, stream>>>(w2, w2_bf, nf);
  }

  // Embedding + mem prefix -> residual stream h (f32).
  {
    int n = S_ * D_;
    embed_kernel<<<(n + 255) / 256, 256, 0, stream>>>(tokens, emb, pos, mem, h);
  }

  for (int l = 0; l < L_; ++l) {
    const __bf16* Wq = wq_bf + (size_t)l * D_ * D_;
    const __bf16* Wk = wk_bf + (size_t)l * D_ * D_;
    const __bf16* Wv = wv_bf + (size_t)l * D_ * D_;
    const __bf16* Wo = wo_bf + (size_t)l * D_ * D_;
    const __bf16* W1 = w1_bf + (size_t)l * FF_ * D_;
    const __bf16* W2 = w2_bf + (size_t)l * FF_ * D_;

    // LN1 -> bf16 activations
    ln_bf16_kernel<<<S_, 256, 0, stream>>>(h, nrm, ln1_s + l * D_, ln1_b + l * D_);

    // Q, K projections (bf16 out); V projection stored transposed vT[D,S]
    gemm(nrm, D_, Wq, D_, nullptr, qb, D_, nullptr, nullptr, S_, D_, D_, EPI_BF16, 0, 1, 0, 0, 0);
    gemm(nrm, D_, Wk, D_, nullptr, kb, D_, nullptr, nullptr, S_, D_, D_, EPI_BF16, 0, 1, 0, 0, 0);
    gemm(nrm, D_, Wv, D_, nullptr, vT, S_, nullptr, nullptr, S_, D_, D_, EPI_BF16_TRANS, 0, 1, 0, 0, 0);

    // Scores + softmax per head (scores buffer reused; probs kept per head).
    for (int hh = 0; hh < H_; ++hh) {
      // scores[q,k] = Qh[q,:] . Kh[k,:]  (A.W^T form, K=64), causal block skip
      gemm(qb + hh * HD_, D_, kb + hh * HD_, D_, scores, nullptr, S_,
           nullptr, nullptr, S_, S_, HD_, EPI_F32, 1, 1, 0, 0, 0);
      softmax_causal_kernel<<<S_, 256, 0, stream>>>(scores, P + (size_t)hh * S_ * S_);
    }
    // Batched P.V over all heads: z = head (12 x 66 waves in one launch).
    // O_h[q,d] = sum_k P_h[q,k] * vT[h*64+d, k]  (A.W^T form, K=S)
    gemm(P, S_, vT, S_, nullptr, ob, D_, nullptr, nullptr,
         S_, HD_, S_, EPI_BF16, 0,
         H_, (size_t)S_ * S_, (size_t)HD_ * S_, (size_t)HD_);

    // Output projection + residual (h += ob . Wo^T)
    gemm(ob, D_, Wo, D_, h, nullptr, D_, nullptr, h, S_, D_, D_, EPI_F32_RES, 0, 1, 0, 0, 0);

    // LN2 -> MLP
    ln_bf16_kernel<<<S_, 256, 0, stream>>>(h, nrm, ln2_s + l * D_, ln2_b + l * D_);
    gemm(nrm, D_, W1, D_, nullptr, act1, FF_, b1 + (size_t)l * FF_, nullptr,
         S_, FF_, D_, EPI_BF16_GELU, 0, 1, 0, 0, 0);
    gemm(act1, FF_, W2, FF_, h, nullptr, D_, b2 + (size_t)l * D_, h,
         S_, D_, FF_, EPI_F32_RES, 0, 1, 0, 0, 0);
  }

  // Final LN over the last M rows -> d_out (f32 [64, 768])
  ln_f32out_kernel<<<M_, 256, 0, stream>>>(h + (size_t)(S_ - M_) * D_, out, lnm_s, lnm_b);
}
